// T5Attention_78520592105552
// MI455X (gfx1250) — compile-verified, matrix operand order checked by reading, not run on hardware
//
#include <hip/hip_runtime.h>
#include <hip/hip_bf16.h>
#include <math.h>

// Problem constants (match reference)
#define BB      4
#define SS      2048
#define DD      512
#define NH      8
#define DH      64
#define NBUCK   32
#define MAXDIST 128

typedef __attribute__((ext_vector_type(16))) __bf16   bf16x16;
typedef __attribute__((ext_vector_type(8)))  __bf16   bf16x8;
typedef __attribute__((ext_vector_type(8)))  float    f32x8;
typedef __attribute__((ext_vector_type(4)))  uint32_t u32x4;
typedef __attribute__((ext_vector_type(8)))  uint32_t u32x8;

__device__ __forceinline__ f32x8 zero8() {
  f32x8 z;
#pragma unroll
  for (int e = 0; e < 8; ++e) z[e] = 0.0f;
  return z;
}

__device__ __forceinline__ f32x8 wmma_bf16(bf16x16 a, bf16x16 b, f32x8 c) {
  // 8 args: (neg_a, A, neg_b, B, c_mod, C, reuse_a, reuse_b)
  return __builtin_amdgcn_wmma_f32_16x16x32_bf16(
      false, a, false, b, (short)0, c, false, false);
}

__device__ __forceinline__ bf16x16 join8(bf16x8 lo, bf16x8 hi) {
  bf16x16 r;
#pragma unroll
  for (int e = 0; e < 8; ++e) { r[e] = lo[e]; r[e + 8] = hi[e]; }
  return r;
}

// Load a 16x32 bf16 fragment in the ISA's 16-bit A layout from a row-major
// source (row stride `stride` elements). Lane L holds row M = L&15; lanes
// 0-15 cover K=0..7 / 16..23, lanes 16-31 cover K=8..15 / 24..31.
// Works on global or LDS pointers (addrspace inferred after inlining).
__device__ __forceinline__ bf16x16 load_frag(const __bf16* __restrict__ base,
                                             int stride, int lane) {
  const int m  = lane & 15;
  const int kb = (lane >> 4) << 3;        // 0 or 8
  const __bf16* p = base + (size_t)m * stride + kb;
  bf16x8 lo = *(const bf16x8*)(p);        // K = kb .. kb+7      (16B aligned)
  bf16x8 hi = *(const bf16x8*)(p + 16);   // K = 16+kb .. 16+kb+7
  return join8(lo, hi);
}

// ------------------------------------------------------ Tensor Data Mover
// Issue a 2D TENSOR_LOAD_TO_LDS per CDNA5 ISA.
// g0: count=1 | lds_addr | global_addr[56:0] | type=2
// g1: workgroup_mask=0, data_size=1 (2B), dims/strides packed per spec.
__device__ __forceinline__ void tdm_load_2d(uint32_t ldsOff, const void* gptr,
                                            u32x8 g1) {
  uint64_t ga = (uint64_t)(uintptr_t)gptr;
  u32x4 g0;
  g0[0] = 1u;                                               // count = 1
  g0[1] = ldsOff;                                           // lds_addr
  g0[2] = (uint32_t)ga;                                     // global_addr lo
  g0[3] = ((uint32_t)(ga >> 32) & 0x01FFFFFFu) | 0x80000000u; // hi | type=2
  asm volatile("tensor_load_to_lds %0, %1, null, null"
               :: "s"(g0), "s"(g1) : "memory");
}

// Pack group1 for a 2D tile: tile_dim0 x tile_dim1 of 2-byte elements,
// row stride `stride0` (data_size units), tensor dims sized to the tile.
__device__ __forceinline__ u32x8 tdm_group1_2d(uint32_t tensor_d0,
                                               uint32_t tensor_d1,
                                               uint32_t tile_d0,
                                               uint32_t tile_d1,
                                               uint32_t stride0) {
  u32x8 g1;
  g1[0] = 0x00010000u;                          // mask=0, data_size=1 (2B)
  g1[1] = (tensor_d0 & 0xFFFFu) << 16;          // tensor_dim0[15:0] @ bit48
  g1[2] = (tensor_d0 >> 16) | ((tensor_d1 & 0xFFFFu) << 16);
  g1[3] = (tensor_d1 >> 16) | (tile_d0 << 16);  // tile_dim0 @ bit112
  g1[4] = tile_d1;                              // tile_dim1 @ bit128, dim2=0
  g1[5] = stride0;                              // tensor_dim0_stride lo
  g1[6] = 0u;                                   // stride hi, dim1_stride lo
  g1[7] = 0u;
  return g1;
}

// ---------------------------------------------------------------- converts
__global__ void cvt_bf16_kernel(const float* __restrict__ src,
                                __bf16* __restrict__ dst, int n) {
  int i = blockIdx.x * blockDim.x + threadIdx.x;
  if (i < n) dst[i] = (__bf16)src[i];
}

// Wt[o][k] = W[k][o]  (store W transposed in bf16)
__global__ void cvt_wt_kernel(const float* __restrict__ W,
                              __bf16* __restrict__ Wt) {
  int i = blockIdx.x * blockDim.x + threadIdx.x;
  if (i >= DD * DD) return;
  int o = i / DD, k = i % DD;
  Wt[i] = (__bf16)W[k * DD + o];
}

// bias table tab[h][delta + S-1], delta = k - q in [-(S-1), S-1]
__global__ void bias_kernel(const float* __restrict__ rel_bias,
                            float* __restrict__ tab) {
  const int W = 2 * SS - 1;
  int i = blockIdx.x * blockDim.x + threadIdx.x;
  if (i >= NH * W) return;
  int h   = i / W;
  int rel = (i % W) - (SS - 1);
  const int nb = NBUCK / 2;        // 16 (bidirectional)
  int bucket = (rel > 0) ? nb : 0;
  int rp = rel < 0 ? -rel : rel;
  const int max_exact = nb / 2;    // 8
  int bsmall;
  if (rp < max_exact) {
    bsmall = rp;
  } else {
    float t = __logf((float)rp / (float)max_exact) /
              __logf((float)MAXDIST / (float)max_exact) *
              (float)(nb - max_exact);
    bsmall = max_exact + (int)t;
    if (bsmall > nb - 1) bsmall = nb - 1;
  }
  tab[i] = rel_bias[(bucket + bsmall) * NH + h];
}

// ---------------------------------------------------------------- GEMM
// One wave computes a 16x64 tile of Y = A[M,512] @ W[512,512] (Wt = W^T).
// Manually 2x-unrolled ping-pong pipeline: P-set and Q-set fragments are
// loaded and consumed in alternating positions so no loop-carried register
// copies are needed, and loads stay in flight across the WMMAs.
// mode 0: Y -> bf16 [B,H,S,64]    (Q, K)
// mode 1: Y -> bf16 [B,H,64,S]    (V transposed)
// mode 2: Y -> f32  [M, 512]      (final output)
__global__ __launch_bounds__(128) void gemm16_kernel(
    const __bf16* __restrict__ A, const __bf16* __restrict__ Wt,
    void* __restrict__ out, int mode) {
  const int M    = BB * SS;
  const int lane = threadIdx.x & 31;
  const int wave = blockIdx.x * (blockDim.x >> 5) + (threadIdx.x >> 5);
  const int colTiles = DD / 64;                 // 8
  const int rowTile  = wave / colTiles;
  const int colTile  = wave % colTiles;
  if (rowTile >= M / 16) return;                // wave-uniform
  const int row0 = rowTile * 16;
  const int col0 = colTile * 64;

  const __bf16* Arow = A + (size_t)row0 * DD;

  f32x8 acc[4];
#pragma unroll
  for (int f = 0; f < 4; ++f) acc[f] = zero8();

  bf16x16 aP = load_frag(Arow, DD, lane);
  bf16x16 bP[4];
#pragma unroll
  for (int f = 0; f < 4; ++f)
    bP[f] = load_frag(Wt + (size_t)(col0 + f * 16) * DD, DD, lane);

  bf16x16 aQ, bQ[4];
  for (int kk = 0; kk + 64 < DD; kk += 64) {
    __builtin_prefetch(Arow + kk + 96, 0, 3);
    // load step kk+32 into Q-set (in flight across P-set WMMAs)
    aQ = load_frag(Arow + kk + 32, DD, lane);
#pragma unroll
    for (int f = 0; f < 4; ++f)
      bQ[f] = load_frag(Wt + (size_t)(col0 + f * 16) * DD + kk + 32, DD, lane);
#pragma unroll
    for (int f = 0; f < 4; ++f) acc[f] = wmma_bf16(aP, bP[f], acc[f]);
    // load step kk+64 into P-set (in flight across Q-set WMMAs)
    aP = load_frag(Arow + kk + 64, DD, lane);
#pragma unroll
    for (int f = 0; f < 4; ++f)
      bP[f] = load_frag(Wt + (size_t)(col0 + f * 16) * DD + kk + 64, DD, lane);
#pragma unroll
    for (int f = 0; f < 4; ++f) acc[f] = wmma_bf16(aQ, bQ[f], acc[f]);
  }
  // epilogue: steps DD-64 (P-set, already loaded) and DD-32 (Q-set)
  aQ = load_frag(Arow + DD - 32, DD, lane);
#pragma unroll
  for (int f = 0; f < 4; ++f)
    bQ[f] = load_frag(Wt + (size_t)(col0 + f * 16) * DD + DD - 32, DD, lane);
#pragma unroll
  for (int f = 0; f < 4; ++f) acc[f] = wmma_bf16(aP, bP[f], acc[f]);
#pragma unroll
  for (int f = 0; f < 4; ++f) acc[f] = wmma_bf16(aQ, bQ[f], acc[f]);

  // C layout: lane L holds col n=L&15; VGPR r holds row r (lanes 0-15) or
  // r+8 (lanes 16-31).
  const int n  = lane & 15;
  const int hl = lane >> 4;
#pragma unroll
  for (int f = 0; f < 4; ++f) {
#pragma unroll
    for (int r = 0; r < 8; ++r) {
      int grow = row0 + r + 8 * hl;
      int gcol = col0 + f * 16 + n;
      float v  = acc[f][r];
      if (mode == 2) {
        ((float*)out)[(size_t)grow * DD + gcol] = v;
      } else {
        int b_ = grow / SS, s_ = grow % SS;
        int h_ = gcol / DH, d_ = gcol % DH;
        if (mode == 0)
          ((__bf16*)out)[(((size_t)(b_ * NH + h_) * SS) + s_) * DH + d_] =
              (__bf16)v;
        else
          ((__bf16*)out)[(((size_t)(b_ * NH + h_) * DH) + d_) * SS + s_] =
              (__bf16)v;
      }
    }
  }
}

// ---------------------------------------------------------------- attention
#define QTPB 4   // q-tiles (waves) per block; all share (b,h) and K/V tiles

__global__ __launch_bounds__(128) void attn_kernel(
    const __bf16* __restrict__ Q,       // [B,H,S,64]
    const __bf16* __restrict__ K,       // [B,H,S,64]
    const __bf16* __restrict__ Vt,      // [B,H,64,S]
    const float* __restrict__ biasTab,  // [H, 2S-1]
    __bf16* __restrict__ ctx) {         // [B,S,512]
  __shared__ float biasS[2 * SS - 1];               // 16380 B (one head/block)
  __shared__ float pT[QTPB][16][33];                // P transpose staging
  __shared__ alignas(16) __bf16 kStage[2][32 * DH]; // 2x K tile [32k][64d]
  __shared__ alignas(16) __bf16 vStage[2][DH * 32]; // 2x V tile [64d][32k]

  const int lane  = threadIdx.x & 31;
  const int warp  = threadIdx.x >> 5;
  const int bh    = blockIdx.x;
  const int b     = bh / NH, h = bh % NH;
  const int qBase = (blockIdx.y * QTPB + warp) * 16;

  for (int i = threadIdx.x; i < 2 * SS - 1; i += blockDim.x)
    biasS[i] = biasTab[h * (2 * SS - 1) + i];

  const __bf16* Qb = Q  + (size_t)bh * SS * DH;
  const __bf16* Kb = K  + (size_t)bh * SS * DH;
  const __bf16* Vb = Vt + (size_t)bh * DH * SS;

  // TDM descriptors (loop-invariant parts). K tile: 64x32 rows; V tile:
  // 32-wide slice of 64 rows with row stride S.
  const uint32_t kOff0 = (uint32_t)(uintptr_t)&kStage[0][0];
  const uint32_t vOff0 = (uint32_t)(uintptr_t)&vStage[0][0];
  const uint32_t kBufB = 32 * DH * 2;   // bytes per K buffer
  const uint32_t vBufB = DH * 32 * 2;   // bytes per V buffer
  const u32x8 g1K = tdm_group1_2d(/*tensor_d0=*/DH, /*tensor_d1=*/32,
                                  /*tile_d0=*/DH, /*tile_d1=*/32,
                                  /*stride0=*/DH);
  const u32x8 g1V = tdm_group1_2d(/*tensor_d0=*/32, /*tensor_d1=*/DH,
                                  /*tile_d0=*/32, /*tile_d1=*/DH,
                                  /*stride0=*/SS);

  bf16x16 aQ0 = load_frag(Qb + (size_t)qBase * DH,      DH, lane); // d 0..31
  bf16x16 aQ1 = load_frag(Qb + (size_t)qBase * DH + 32, DH, lane); // d 32..63

  float mrow[8], lrow[8];
  f32x8 oacc[4];
#pragma unroll
  for (int r = 0; r < 8; ++r) { mrow[r] = -INFINITY; lrow[r] = 0.0f; }
#pragma unroll
  for (int f = 0; f < 4; ++f) oacc[f] = zero8();

  const int n  = lane & 15;
  const int hl = lane >> 4;
  const int biasBase = (SS - 1) - qBase;     // table index = (k - q) + S-1

  // prologue DMA: tile 0 -> buffer 0
  if (warp == 0) {
    tdm_load_2d(kOff0, Kb, g1K);
    tdm_load_2d(vOff0, Vb, g1V);
    __builtin_amdgcn_s_wait_tensorcnt(0);
  }
  __syncthreads();   // bias table + tile 0 visible to all waves

  for (int k0 = 0; k0 < SS; k0 += 32) {
    const uint32_t cur = (uint32_t)(k0 >> 5) & 1u;
    const uint32_t nxt = cur ^ 1u;

    // kick off DMA for tile t+1 into the other buffer; it runs on the TDM
    // engine while this tile's WMMAs + softmax execute.
    if (warp == 0 && k0 + 32 < SS) {
      tdm_load_2d(kOff0 + nxt * kBufB, Kb + (size_t)(k0 + 32) * DH, g1K);
      tdm_load_2d(vOff0 + nxt * vBufB, Vb + (k0 + 32), g1V);
    }
    const __bf16* kS = kStage[cur];
    const __bf16* vS = vStage[cur];

    // scores S[16q,32k] = Q[16,64] * K^T (row-major K => A-style frags)
    bf16x16 bK0a = load_frag(kS,                DH, lane);
    bf16x16 bK0b = load_frag(kS + 32,           DH, lane);
    bf16x16 bK1a = load_frag(kS + 16 * DH,      DH, lane);
    bf16x16 bK1b = load_frag(kS + 16 * DH + 32, DH, lane);

    f32x8 s0 = zero8(), s1 = zero8();
    s0 = wmma_bf16(aQ0, bK0a, s0);
    s0 = wmma_bf16(aQ1, bK0b, s0);
    s1 = wmma_bf16(aQ0, bK1a, s1);
    s1 = wmma_bf16(aQ1, bK1b, s1);

    // relative-position bias (LDS table)
#pragma unroll
    for (int r = 0; r < 8; ++r) {
      int idx = biasBase - (r + 8 * hl) + k0 + n;
      s0[r] += biasS[idx];
      s1[r] += biasS[idx + 16];
    }

    // online softmax: rows live across 16-lane halves -> xor 1/2/4/8 reduce
    float p0[8], p1[8], corr[8];
#pragma unroll
    for (int r = 0; r < 8; ++r) {
      float t = fmaxf(s0[r], s1[r]);
      t = fmaxf(t, __shfl_xor(t, 1, 32));
      t = fmaxf(t, __shfl_xor(t, 2, 32));
      t = fmaxf(t, __shfl_xor(t, 4, 32));
      t = fmaxf(t, __shfl_xor(t, 8, 32));
      float mnew = fmaxf(mrow[r], t);
      p0[r]   = __expf(s0[r] - mnew);
      p1[r]   = __expf(s1[r] - mnew);
      corr[r] = __expf(mrow[r] - mnew);
      float rs = p0[r] + p1[r];
      rs += __shfl_xor(rs, 1, 32);
      rs += __shfl_xor(rs, 2, 32);
      rs += __shfl_xor(rs, 4, 32);
      rs += __shfl_xor(rs, 8, 32);
      lrow[r] = lrow[r] * corr[r] + rs;
      mrow[r] = mnew;
    }
#pragma unroll
    for (int f = 0; f < 4; ++f)
#pragma unroll
      for (int r = 0; r < 8; ++r) oacc[f][r] *= corr[r];

    // transpose P (C layout) -> A layout via per-wave LDS tile
#pragma unroll
    for (int r = 0; r < 8; ++r) {
      pT[warp][r + 8 * hl][n]      = p0[r];
      pT[warp][r + 8 * hl][16 + n] = p1[r];
    }
    bf16x16 pA;
    {
      const int m  = lane & 15;
      const int kb = hl << 3;
#pragma unroll
      for (int e = 0; e < 8; ++e) {
        pA[e]     = (__bf16)pT[warp][m][kb + e];
        pA[e + 8] = (__bf16)pT[warp][m][16 + kb + e];
      }
    }

    // O[16q,64d] += P[16,32k] * V[32k,64d]  (Vt pre-transposed => A-style)
#pragma unroll
    for (int f = 0; f < 4; ++f) {
      bf16x16 bV = load_frag(vS + (size_t)(f * 16) * 32, 32, lane);
      oacc[f] = wmma_bf16(pA, bV, oacc[f]);
    }

    // tile t+1 landed; barrier also fences buffer reuse across waves
    if (warp == 0) __builtin_amdgcn_s_wait_tensorcnt(0);
    __syncthreads();
  }

#pragma unroll
  for (int r = 0; r < 8; ++r) lrow[r] = 1.0f / lrow[r];
#pragma unroll
  for (int f = 0; f < 4; ++f) {
#pragma unroll
    for (int r = 0; r < 8; ++r) {
      int s_ = qBase + r + 8 * hl;
      int c_ = h * DH + f * 16 + n;
      ctx[((size_t)b * SS + s_) * DD + c_] = (__bf16)(oacc[f][r] * lrow[r]);
    }
  }
}

// ---------------------------------------------------------------- launch
extern "C" void kernel_launch(void* const* d_in, const int* in_sizes, int n_in,
                              void* d_out, int out_size, void* d_ws,
                              size_t ws_size, hipStream_t stream) {
  (void)in_sizes; (void)n_in; (void)out_size; (void)ws_size;
  const float* hidden = (const float*)d_in[0];
  const float* Wq = (const float*)d_in[1];
  const float* Wk = (const float*)d_in[2];
  const float* Wv = (const float*)d_in[3];
  const float* Wo = (const float*)d_in[4];
  const float* rb = (const float*)d_in[5];

  char* ws = (char*)d_ws;
  size_t off = 0;
  auto alloc = [&](size_t bytes) -> void* {
    void* p = ws + off;
    off += (bytes + 255) & ~(size_t)255;
    return p;
  };
  __bf16* Xb   = (__bf16*)alloc((size_t)BB * SS * DD * 2);   // 8 MiB
  __bf16* Wqb  = (__bf16*)alloc((size_t)DD * DD * 2);
  __bf16* Wkb  = (__bf16*)alloc((size_t)DD * DD * 2);
  __bf16* Wvb  = (__bf16*)alloc((size_t)DD * DD * 2);
  __bf16* Wob  = (__bf16*)alloc((size_t)DD * DD * 2);
  __bf16* Qb   = (__bf16*)alloc((size_t)BB * NH * SS * DH * 2);
  __bf16* Kb   = (__bf16*)alloc((size_t)BB * NH * SS * DH * 2);
  __bf16* Vtb  = (__bf16*)alloc((size_t)BB * NH * DH * SS * 2);
  __bf16* Ctx  = (__bf16*)alloc((size_t)BB * SS * DD * 2);
  float*  bias = (float*)alloc((size_t)NH * (2 * SS - 1) * 4);

  const int nX = BB * SS * DD;
  cvt_bf16_kernel<<<(nX + 255) / 256, 256, 0, stream>>>(hidden, Xb, nX);
  const int nW = DD * DD;
  cvt_wt_kernel<<<(nW + 255) / 256, 256, 0, stream>>>(Wq, Wqb);
  cvt_wt_kernel<<<(nW + 255) / 256, 256, 0, stream>>>(Wk, Wkb);
  cvt_wt_kernel<<<(nW + 255) / 256, 256, 0, stream>>>(Wv, Wvb);
  cvt_wt_kernel<<<(nW + 255) / 256, 256, 0, stream>>>(Wo, Wob);
  const int nB = NH * (2 * SS - 1);
  bias_kernel<<<(nB + 255) / 256, 256, 0, stream>>>(rb, bias);

  // (8192/16 row tiles) * (512/64 col tiles) waves, 4 waves per block
  const int gemmBlocks = (BB * SS / 16) * (DD / 64) / 4;   // 1024
  gemm16_kernel<<<gemmBlocks, 128, 0, stream>>>(Xb, Wqb, Qb, 0);
  gemm16_kernel<<<gemmBlocks, 128, 0, stream>>>(Xb, Wkb, Kb, 0);
  gemm16_kernel<<<gemmBlocks, 128, 0, stream>>>(Xb, Wvb, Vtb, 1);

  dim3 ag(BB * NH, SS / (16 * QTPB));                      // (32, 32)
  attn_kernel<<<ag, 128, 0, stream>>>(Qb, Kb, Vtb, bias, Ctx);

  gemm16_kernel<<<gemmBlocks, 128, 0, stream>>>(Ctx, Wob, d_out, 2);
}
